// HeteroSAGE_46858093199984
// MI455X (gfx1250) — compile-verified, matrix operand order checked by reading, not run on hardware
//
#include <hip/hip_runtime.h>
#include <hip/hip_bf16.h>

// ---------------------------------------------------------------------------
// CDNA5 (gfx1250) HeteroGNN forward.
// Conv-layer GEMMs: bf16 A staged in memory (fits 192MB L2), B pre-packed to
// WMMA fragment order; software-pipelined inner loop: stage t+1 b128 loads
// issued before stage t's 4x v_wmma_f32_16x16x32_bf16 (partial loadcnt waits).
// Sparse segment ops: hw float atomics + wave32 shuffles.
// ---------------------------------------------------------------------------

typedef __attribute__((ext_vector_type(16))) __bf16 v16bf;
typedef __attribute__((ext_vector_type(8)))  float  v8f;

#define DEVINL __device__ __forceinline__

DEVINL unsigned short f2bfbits(float f) {
  unsigned u = __builtin_bit_cast(unsigned, f);
  unsigned r = u + 0x7FFFu + ((u >> 16) & 1u);   // round-to-nearest-even
  return (unsigned short)(r >> 16);
}
DEVINL __bf16 f2bf(float f) {
  unsigned short h = f2bfbits(f);
  return __builtin_bit_cast(__bf16, h);
}

DEVINL void atomAddF(float* p, float v) {
  __hip_atomic_fetch_add(p, v, __ATOMIC_RELAXED, __HIP_MEMORY_SCOPE_AGENT);
}

// Monotone order-preserving float <-> uint key (atomicMax-based segment max)
DEVINL unsigned fkey(float f) {
  unsigned u = __builtin_bit_cast(unsigned, f);
  return (u & 0x80000000u) ? ~u : (u | 0x80000000u);
}
DEVINL float fdecode(unsigned k) {
  unsigned u = (k & 0x80000000u) ? (k ^ 0x80000000u) : ~k;
  return __builtin_bit_cast(float, u);
}

// ---------------------------------------------------------------------------
// Pack W[K,128] fp32 -> bf16 in WMMA B-fragment order (zero padded to K32*32).
// Element (t, nt, lane, j=2v+s) <- W[k, nt*16 + lane%16],
//   k = t*32 + (v<4?0:16) + (lane/16)*8 + 2*(v&3) + s
// ---------------------------------------------------------------------------
__global__ void k_pack_w(const float* __restrict__ W, unsigned short* __restrict__ P,
                         int K, int total)
{
  int gid = blockIdx.x * blockDim.x + threadIdx.x;
  if (gid >= total) return;
  int j  = gid & 15;
  int l  = (gid >> 4) & 31;
  int nt = (gid >> 9) & 7;
  int t  = gid >> 12;
  int v  = j >> 1, s = j & 1;
  int k  = t * 32 + ((v < 4) ? 0 : 16) + (l >> 4) * 8 + 2 * (v & 3) + s;
  int col = nt * 16 + (l & 15);
  float val = (k < K) ? W[(size_t)k * 128 + col] : 0.f;
  P[gid] = f2bfbits(val);
}

// ---------------------------------------------------------------------------
// bf16-A WMMA GEMM (K == 128): Y[M,128] = op(Abf[M,128] @ W + bias).
// One wave computes 64x16; 4 K-steps, double-buffered: next-step loads are
// issued before current-step WMMAs so the wave overlaps VMEM with matrix ops.
// ---------------------------------------------------------------------------
__global__ __launch_bounds__(32)
void k_gemm_bf(const unsigned short* __restrict__ A,   // [M,128] bf16 row-major
               const unsigned short* __restrict__ P,   // packed B fragments
               const float* __restrict__ bias, float* __restrict__ Y,
               int M, int accum, int relu)
{
  const int lane  = threadIdx.x & 31;
  const int m16   = lane & 15;
  const int khalf = lane >> 4;
  const int nt    = blockIdx.y;
  const int col   = nt * 16 + m16;
  const int mbase = blockIdx.x * 64;

  const uint4* pa[4];
#pragma unroll
  for (int u = 0; u < 4; ++u) {
    int rr = mbase + u * 16 + m16;
    rr = rr < M ? rr : (M - 1);
    pa[u] = (const uint4*)(A + (size_t)rr * 128);
  }

  v8f c[4];
#pragma unroll
  for (int u = 0; u < 4; ++u) c[u] = v8f{0.f,0.f,0.f,0.f,0.f,0.f,0.f,0.f};

  const v16bf* __restrict__ PB = (const v16bf*)P;

  // stage 0 loads
  v16bf bcur = PB[(size_t)nt * 32 + lane];
  uint4 acur[4][2];
  {
    int c0 = khalf;                       // (khalf*8)>>3
#pragma unroll
    for (int u = 0; u < 4; ++u) { acur[u][0] = pa[u][c0]; acur[u][1] = pa[u][c0 + 2]; }
  }

#pragma unroll
  for (int t = 0; t < 4; ++t) {
    v16bf bnxt;
    uint4 anxt[4][2];
    if (t < 3) {                          // issue stage t+1 loads first
      bnxt = PB[((size_t)(t + 1) * 8 + nt) * 32 + lane];
      int c0 = ((t + 1) * 32 + khalf * 8) >> 3;
#pragma unroll
      for (int u = 0; u < 4; ++u) { anxt[u][0] = pa[u][c0]; anxt[u][1] = pa[u][c0 + 2]; }
    }
#pragma unroll
    for (int u = 0; u < 4; ++u) {
      union { v16bf v; uint4 q[2]; } af;
      af.q[0] = acur[u][0];
      af.q[1] = acur[u][1];
      c[u] = __builtin_amdgcn_wmma_f32_16x16x32_bf16(false, af.v, false, bcur,
                                                     (short)0, c[u], false, false);
    }
    if (t < 3) {                          // renamed away by full unroll
      bcur = bnxt;
#pragma unroll
      for (int u = 0; u < 4; ++u) { acur[u][0] = anxt[u][0]; acur[u][1] = anxt[u][1]; }
    }
  }

  float bc = bias ? bias[col] : 0.f;
#pragma unroll
  for (int u = 0; u < 4; ++u) {
#pragma unroll
    for (int rr = 0; rr < 8; ++rr) {
      int orow = mbase + u * 16 + rr + 8 * khalf;
      if (orow < M) {
        float v = c[u][rr] + bc;
        size_t o = (size_t)orow * 128 + col;
        if (accum) v += Y[o];
        if (relu)  v = v > 0.f ? v : 0.f;
        Y[o] = v;
      }
    }
  }
}

// ---------------------------------------------------------------------------
// fp32-A WMMA GEMM (projection only, K % 8 == 0): converts A in-register,
// writes fp32 Y and optional bf16 copy Ybf.
// ---------------------------------------------------------------------------
__global__ __launch_bounds__(32)
void k_gemm_f32(const float* __restrict__ A, const unsigned short* __restrict__ P,
                const float* __restrict__ bias, float* __restrict__ Y,
                unsigned short* __restrict__ Ybf, int M, int K, int relu)
{
  const int lane  = threadIdx.x & 31;
  const int m16   = lane & 15;
  const int khalf = lane >> 4;
  const int nt    = blockIdx.y;
  const int col   = nt * 16 + m16;
  const int mbase = blockIdx.x * 64;
  const int K32   = (K + 31) >> 5;

  int r[4];
#pragma unroll
  for (int u = 0; u < 4; ++u) {
    int rr = mbase + u * 16 + m16;
    r[u] = rr < M ? rr : (M - 1);
  }

  v8f c[4];
#pragma unroll
  for (int u = 0; u < 4; ++u) c[u] = v8f{0.f,0.f,0.f,0.f,0.f,0.f,0.f,0.f};

  const v16bf* __restrict__ PB = (const v16bf*)P;

  for (int t = 0; t < K32; ++t) {
    v16bf b = PB[((size_t)t * 8 + nt) * 32 + lane];
    int c0s = t * 32 + khalf * 8;
    int c1s = c0s + 16;
    float s0 = (c0s + 8 <= K) ? 1.f : 0.f;
    float s1 = (c1s + 8 <= K) ? 1.f : 0.f;
    int cc0 = (c0s + 8 <= K) ? c0s : (K - 8);
    int cc1 = (c1s + 8 <= K) ? c1s : (K - 8);
#pragma unroll
    for (int u = 0; u < 4; ++u) {
      const float4* pa = (const float4*)(A + (size_t)r[u] * K);
      float4 f0 = pa[(cc0 >> 2)];
      float4 f1 = pa[(cc0 >> 2) + 1];
      float4 f2 = pa[(cc1 >> 2)];
      float4 f3 = pa[(cc1 >> 2) + 1];
      v16bf a;
      a[0]  = f2bf(f0.x * s0); a[1]  = f2bf(f0.y * s0);
      a[2]  = f2bf(f0.z * s0); a[3]  = f2bf(f0.w * s0);
      a[4]  = f2bf(f1.x * s0); a[5]  = f2bf(f1.y * s0);
      a[6]  = f2bf(f1.z * s0); a[7]  = f2bf(f1.w * s0);
      a[8]  = f2bf(f2.x * s1); a[9]  = f2bf(f2.y * s1);
      a[10] = f2bf(f2.z * s1); a[11] = f2bf(f2.w * s1);
      a[12] = f2bf(f3.x * s1); a[13] = f2bf(f3.y * s1);
      a[14] = f2bf(f3.z * s1); a[15] = f2bf(f3.w * s1);
      c[u] = __builtin_amdgcn_wmma_f32_16x16x32_bf16(false, a, false, b,
                                                     (short)0, c[u], false, false);
    }
  }

  float bc = bias ? bias[col] : 0.f;
#pragma unroll
  for (int u = 0; u < 4; ++u) {
#pragma unroll
    for (int rr = 0; rr < 8; ++rr) {
      int orow = mbase + u * 16 + rr + 8 * khalf;
      if (orow < M) {
        float v = c[u][rr] + bc;
        if (relu) v = v > 0.f ? v : 0.f;
        size_t o = (size_t)orow * 128 + col;
        Y[o] = v;
        if (Ybf) Ybf[o] = f2bfbits(v);
      }
    }
  }
}

// ---------------------------------------------------------------------------
// Elementwise / scatter kernels
// ---------------------------------------------------------------------------
__global__ void k_zero(float* __restrict__ p, long long n) {
  long long i = (long long)blockIdx.x * blockDim.x + threadIdx.x;
  long long st = (long long)gridDim.x * blockDim.x;
  for (; i < n; i += st) p[i] = 0.f;
}

__global__ void k_add_bias(float* __restrict__ acc, const float* __restrict__ bias,
                           long long n) {
  long long i = (long long)blockIdx.x * blockDim.x + threadIdx.x;
  long long st = (long long)gridDim.x * blockDim.x;
  for (; i < n; i += st) acc[i] += bias[i & 127];
}

// mean = sum / max(cnt,1); writes fp32 in place + bf16 copy
__global__ void k_mean(float* __restrict__ sum, const float* __restrict__ cnt,
                       unsigned short* __restrict__ sumbf, long long n) {
  long long i = (long long)blockIdx.x * blockDim.x + threadIdx.x;
  long long st = (long long)gridDim.x * blockDim.x;
  for (; i < n; i += st) {
    float v = sum[i] / fmaxf(cnt[i >> 7], 1.f);
    sum[i] = v;
    sumbf[i] = f2bfbits(v);
  }
}

// GATv2 edge logits: m = leaky_relu(xl[s] + xr[d] + ea@We); logit = m . att
__global__ __launch_bounds__(256)
void k_edge_logit(const float* __restrict__ xl, const float* __restrict__ xr,
                  const int* __restrict__ si, const int* __restrict__ di,
                  const float* __restrict__ ea, const float* __restrict__ We,
                  const float* __restrict__ att, float* __restrict__ logit,
                  unsigned* __restrict__ mx, int E, int ed)
{
  int w = blockIdx.x * (blockDim.x >> 5) + (threadIdx.x >> 5);
  int lane = threadIdx.x & 31;
  if (w >= E) return;
  int s = si[w], d = di[w];
  float acc = 0.f;
#pragma unroll
  for (int j = 0; j < 4; ++j) {
    int c = lane * 4 + j;
    float m = xl[(size_t)s * 128 + c] + xr[(size_t)d * 128 + c];
    for (int t = 0; t < ed; ++t)
      m += ea[(size_t)w * ed + t] * We[t * 128 + c];
    m = m > 0.f ? m : 0.2f * m;                  // leaky_relu(0.2)
    acc += m * att[c];
  }
  for (int o = 16; o; o >>= 1) acc += __shfl_xor(acc, o, 32);
  if (lane == 0) {
    logit[w] = acc;
    atomicMax(mx + d, fkey(acc));
  }
}

__global__ void k_edge_exp(float* __restrict__ logit, const int* __restrict__ di,
                           const unsigned* __restrict__ mx, float* __restrict__ z,
                           int E)
{
  int e = blockIdx.x * blockDim.x + threadIdx.x;
  if (e >= E) return;
  int d = di[e];
  float ex = __expf(logit[e] - fdecode(mx[d]));
  logit[e] = ex;
  atomAddF(z + d, ex);
}

__global__ __launch_bounds__(256)
void k_edge_scatter(const float* __restrict__ xl, const int* __restrict__ si,
                    const int* __restrict__ di, const float* __restrict__ ebuf,
                    const float* __restrict__ z, float* __restrict__ acc, int E)
{
  int w = blockIdx.x * (blockDim.x >> 5) + (threadIdx.x >> 5);
  int lane = threadIdx.x & 31;
  if (w >= E) return;
  int s = si[w], d = di[w];
  float alpha = ebuf[w] / z[d];
#pragma unroll
  for (int j = 0; j < 4; ++j) {
    int c = lane * 4 + j;
    atomAddF(acc + (size_t)d * 128 + c, alpha * xl[(size_t)s * 128 + c]);
  }
}

__global__ __launch_bounds__(256)
void k_sage_scatter(const float* __restrict__ hs, const int* __restrict__ si,
                    const int* __restrict__ di, float* __restrict__ sum,
                    float* __restrict__ cnt, int E)
{
  int w = blockIdx.x * (blockDim.x >> 5) + (threadIdx.x >> 5);
  int lane = threadIdx.x & 31;
  if (w >= E) return;
  int s = si[w], d = di[w];
#pragma unroll
  for (int j = 0; j < 4; ++j) {
    int c = lane + 32 * j;
    atomAddF(sum + (size_t)d * 128 + c, hs[(size_t)s * 128 + c]);
  }
  if (lane == 0) atomAddF(cnt + d, 1.f);
}

// LayerNorm(128) + ReLU, one wave per row; writes fp32 + bf16 copies
__global__ __launch_bounds__(256)
void k_ln_relu(const float* __restrict__ X, const float* __restrict__ g,
               const float* __restrict__ b, float* __restrict__ Y,
               unsigned short* __restrict__ Ybf, int Nrows)
{
  int w = blockIdx.x * (blockDim.x >> 5) + (threadIdx.x >> 5);
  int lane = threadIdx.x & 31;
  if (w >= Nrows) return;
  const float* x = X + (size_t)w * 128;
  float v0 = x[lane], v1 = x[lane + 32], v2 = x[lane + 64], v3 = x[lane + 96];
  float s = v0 + v1 + v2 + v3;
  float q = v0 * v0 + v1 * v1 + v2 * v2 + v3 * v3;
  for (int o = 16; o; o >>= 1) { s += __shfl_xor(s, o, 32); q += __shfl_xor(q, o, 32); }
  float mu  = s * (1.f / 128.f);
  float var = q * (1.f / 128.f) - mu * mu;       // biased, matches torch LN
  float rs  = rsqrtf(var + 1e-5f);
  float* y = Y + (size_t)w * 128;
  unsigned short* yb = Ybf + (size_t)w * 128;
  float o0 = (v0 - mu) * rs * g[lane]      + b[lane];
  float o1 = (v1 - mu) * rs * g[lane + 32] + b[lane + 32];
  float o2 = (v2 - mu) * rs * g[lane + 64] + b[lane + 64];
  float o3 = (v3 - mu) * rs * g[lane + 96] + b[lane + 96];
  o0 = o0 > 0.f ? o0 : 0.f;  o1 = o1 > 0.f ? o1 : 0.f;
  o2 = o2 > 0.f ? o2 : 0.f;  o3 = o3 > 0.f ? o3 : 0.f;
  y[lane] = o0;  y[lane + 32] = o1;  y[lane + 64] = o2;  y[lane + 96] = o3;
  yb[lane]      = f2bfbits(o0);
  yb[lane + 32] = f2bfbits(o1);
  yb[lane + 64] = f2bfbits(o2);
  yb[lane + 96] = f2bfbits(o3);
}

// classifier: out[m] = h[m,:] . W + b
__global__ __launch_bounds__(256)
void k_cls(const float* __restrict__ H, const float* __restrict__ W,
           const float* __restrict__ b, float* __restrict__ out, int M)
{
  int w = blockIdx.x * (blockDim.x >> 5) + (threadIdx.x >> 5);
  int lane = threadIdx.x & 31;
  if (w >= M) return;
  const float* x = H + (size_t)w * 128;
  float acc = x[lane] * W[lane] + x[lane + 32] * W[lane + 32] +
              x[lane + 64] * W[lane + 64] + x[lane + 96] * W[lane + 96];
  for (int o = 16; o; o >>= 1) acc += __shfl_xor(acc, o, 32);
  if (lane == 0) out[w] = acc + b[0];
}

// ---------------------------------------------------------------------------
// Host orchestration
// ---------------------------------------------------------------------------
extern "C" void kernel_launch(void* const* d_in, const int* in_sizes, int n_in,
                              void* d_out, int out_size, void* d_ws, size_t ws_size,
                              hipStream_t stream)
{
  (void)in_sizes; (void)out_size; (void)ws_size;
  static const int NNODES[8] = {2048, 150000, 80000, 200000, 4000, 16000, 16000, 2048};
  static const int FEATS[8]  = {32, 768, 64, 128, 16, 384, 384, 8};
  static const int ES[20]   = {0,2,2,1,0,1,0,3,0,4,0,4,4,2,0,5,0,6,0,7};
  static const int EDd[20]  = {2,0,1,2,1,0,3,0,4,0,4,0,2,4,5,0,6,0,7,0};
  static const int ECNT[20] = {200000,200000,150000,150000,200000,200000,250000,250000,
                               4096,4096,4096,4096,60000,60000,8192,8192,8192,8192,2048,2048};
  static const int EDIM[20] = {4,4,-1,-1,11,11,-1,-1,3,3,3,3,3,3,3,3,3,3,1,1};

  if (n_in < 362) return;

  // ---- decode flattened inputs (setup_inputs() dict insertion order) ----
  int idx = 0;
  const float* X[8];
  for (int t = 0; t < 8; ++t) X[t] = (const float*)d_in[idx++];
  const float* EA[20];
  for (int e = 0; e < 20; ++e) EA[e] = (EDIM[e] > 0) ? (const float*)d_in[idx++] : nullptr;
  const float *projW[8], *projB[8];
  for (int t = 0; t < 8; ++t) { projW[t] = (const float*)d_in[idx++];
                                projB[t] = (const float*)d_in[idx++]; }
  struct CP { const float *Wl, *bl, *Wr, *br, *We, *att, *bias; };
  CP cp[2][20];
  for (int L = 0; L < 2; ++L)
    for (int e = 0; e < 20; ++e) {
      CP p = {};
      p.Wl = (const float*)d_in[idx++];
      p.bl = (const float*)d_in[idx++];
      p.Wr = (const float*)d_in[idx++];
      if (EDIM[e] > 0) {
        p.br   = (const float*)d_in[idx++];
        p.We   = (const float*)d_in[idx++];
        p.att  = (const float*)d_in[idx++];
        p.bias = (const float*)d_in[idx++];
      }
      cp[L][e] = p;
    }
  const float *lng[2][8], *lnb[2][8];
  for (int L = 0; L < 2; ++L)
    for (int t = 0; t < 8; ++t) { lng[L][t] = (const float*)d_in[idx++];
                                  lnb[L][t] = (const float*)d_in[idx++]; }
  const float* clsW = (const float*)d_in[idx++];
  const float* clsB = (const float*)d_in[idx++];
  const int *SRC[20], *DST[20];
  for (int e = 0; e < 20; ++e) SRC[e] = (const int*)d_in[idx++];
  for (int e = 0; e < 20; ++e) DST[e] = (const int*)d_in[idx++];

  // ---- workspace carve (float units, 64-float = 256B aligned blocks) ----
  size_t off[8], tot = 0;
  for (int t = 0; t < 8; ++t) { off[t] = tot; tot += (size_t)NNODES[t]; }
  float* ws = (float*)d_ws;
  size_t p = 0;
  auto carve = [&](size_t nfloats) {
    float* r = ws + p;
    p = (p + nfloats + 63) & ~(size_t)63;
    return r;
  };
  float* h    = carve(tot * 128);
  float* acc  = carve(tot * 128);
  float* tmpA = carve((size_t)200000 * 128);          // xl / SAGE sum
  float* tmpB = carve((size_t)200000 * 128);          // xr
  float* elog = carve(250000);
  float* zbuf = carve(200000);
  unsigned* mx = (unsigned*)carve(200000);
  float* cnt  = carve(200000);
  unsigned short* hbf    = (unsigned short*)carve(tot * 64);            // bf16 h
  unsigned short* tmpAbf = (unsigned short*)carve((size_t)200000 * 64); // bf16 mean
  unsigned short* packW  = (unsigned short*)carve(49152);               // 24*4096 bf16

  auto zgrid = [](long long n) {
    long long bl = (n + 2047) / 2048;
    if (bl < 1) bl = 1; if (bl > 262144) bl = 262144;
    return dim3((unsigned)bl);
  };
  auto pack = [&](const float* W, int K) {
    int total = ((K + 31) >> 5) * 4096;
    k_pack_w<<<dim3((total + 255) / 256), 256, 0, stream>>>(W, packW, K, total);
  };
  auto gemmF = [&](const float* A, const float* W, const float* bias, float* Y,
                   unsigned short* Ybf, int M, int K, int relu) {
    pack(W, K);
    k_gemm_f32<<<dim3((M + 63) / 64, 8), dim3(32), 0, stream>>>(A, packW, bias, Y,
                                                                Ybf, M, K, relu);
  };
  auto gemmB = [&](const unsigned short* Abf, const float* W, const float* bias,
                   float* Y, int M, int accum, int relu) {
    pack(W, 128);
    k_gemm_bf<<<dim3((M + 63) / 64, 8), dim3(32), 0, stream>>>(Abf, packW, bias, Y,
                                                               M, accum, relu);
  };

  // ---- per-type input projection + ReLU (writes h fp32 + bf16) ----
  for (int t = 0; t < 8; ++t)
    gemmF(X[t], projW[t], projB[t], h + off[t] * 128, hbf + off[t] * 128,
          NNODES[t], FEATS[t], 1);

  // ---- two hetero-conv layers ----
  for (int L = 0; L < 2; ++L) {
    k_zero<<<zgrid((long long)tot * 128), 256, 0, stream>>>(acc, (long long)tot * 128);
    for (int e = 0; e < 20; ++e) {
      const CP& P = cp[L][e];
      int s = ES[e], d = EDd[e], E = ECNT[e];
      int Nd = NNODES[d];
      float* accd = acc + off[d] * 128;
      dim3 ge((E + 7) / 8), be(256);
      if (EDIM[e] > 0) {                    // GATv2
        gemmB(hbf + off[s] * 128, P.Wl, P.bl, tmpA, NNODES[s], 0, 0);  // xl
        gemmB(hbf + off[d] * 128, P.Wr, P.br, tmpB, Nd, 0, 0);         // xr
        k_zero<<<zgrid(Nd), 256, 0, stream>>>((float*)mx, Nd);
        k_zero<<<zgrid(Nd), 256, 0, stream>>>(zbuf, Nd);
        k_edge_logit<<<ge, be, 0, stream>>>(tmpA, tmpB, SRC[e], DST[e], EA[e],
                                            P.We, P.att, elog, mx, E, EDIM[e]);
        k_edge_exp<<<dim3((E + 255) / 256), 256, 0, stream>>>(elog, DST[e], mx, zbuf, E);
        k_edge_scatter<<<ge, be, 0, stream>>>(tmpA, SRC[e], DST[e], elog, zbuf, accd, E);
        k_add_bias<<<zgrid((long long)Nd * 128), 256, 0, stream>>>(accd, P.bias,
                                                                   (long long)Nd * 128);
      } else {                              // SAGE (mean aggr)
        k_zero<<<zgrid((long long)Nd * 128), 256, 0, stream>>>(tmpA, (long long)Nd * 128);
        k_zero<<<zgrid(Nd), 256, 0, stream>>>(cnt, Nd);
        k_sage_scatter<<<ge, be, 0, stream>>>(h + off[s] * 128, SRC[e], DST[e],
                                              tmpA, cnt, E);
        k_mean<<<zgrid((long long)Nd * 128), 256, 0, stream>>>(tmpA, cnt, tmpAbf,
                                                               (long long)Nd * 128);
        gemmB(tmpAbf, P.Wl, P.bl, accd, Nd, 1, 0);                // acc += mean@Wl+bl
        gemmB(hbf + off[d] * 128, P.Wr, nullptr, accd, Nd, 1, 0); // acc += xd@Wr
      }
    }
    for (int t = 0; t < 8; ++t) {
      dim3 g((NNODES[t] + 7) / 8), b(256);
      k_ln_relu<<<g, b, 0, stream>>>(acc + off[t] * 128, lng[L][t], lnb[L][t],
                                     h + off[t] * 128, hbf + off[t] * 128, NNODES[t]);
    }
  }

  // ---- classifier on commit nodes ----
  k_cls<<<dim3((2048 + 7) / 8), 256, 0, stream>>>(h + off[0] * 128, clsW, clsB,
                                                  (float*)d_out, 2048);
}
